// TransformerLM_61529701482747
// MI455X (gfx1250) — matrix operand order repaired and checked by reading, not compiled, not run on hardware
//
#include <hip/hip_runtime.h>
#include <cstdint>
#include <cstddef>

// ---------------- types ----------------
typedef __bf16 bf16;
typedef __attribute__((ext_vector_type(16))) __bf16 v16bf;
typedef __attribute__((ext_vector_type(8)))  float  v8f;

// ---------------- problem constants ----------------
constexpr int cV  = 32000;
constexpr int cD  = 1024;
constexpr int cH  = 16;
constexpr int cDh = 64;
constexpr int cF  = 4096;
constexpr int cL  = 12;
constexpr int cS  = 512;
constexpr int cM  = 512;
constexpr int cB  = 4;
constexpr int cT  = cM + cS;      // 1024
constexpr int cHD = cH * cDh;     // 1024
constexpr int cQ  = 3 * cHD;      // 3072
constexpr float cEPS = 1e-5f;

// ---------------- CDNA5 async-copy helpers ----------------
// Per-lane DMA of 16B global -> LDS, tracked by ASYNCcnt (ISA 15.18.3 op 98).
__device__ __forceinline__ uint32_t lds_off32(const void* p) {
    // generic pointer to LDS: low 32 bits are the LDS byte offset
    return (uint32_t)(uintptr_t)p;
}
__device__ __forceinline__ void async_ld_b128(uint32_t lds_dst, const void* gsrc) {
    asm volatile("global_load_async_to_lds_b128 %0, %1, off"
                 :: "v"(lds_dst), "v"(gsrc) : "memory");
}
__device__ __forceinline__ void wait_asynccnt0() {
    asm volatile("s_wait_asynccnt 0" ::: "memory");
}

// ---------------- WMMA GEMM ----------------
// C[M,N](f32) = alpha * A[M,K](bf16) x B[N,K](bf16, row-major "BT" layout),
// batched over blockIdx.z: z -> zb = z/Hcnt, zh = z%Hcnt with separate strides.
// Block tile 256x64x32, 8 waves (4x2), wave tile 64x32 = 8 WMMA per K-step.
// Staging: global_load_async_to_lds_b128, double-buffered.
#define GBM 256
#define GBN 64
#define GBK 32

__global__ __launch_bounds__(256)
void gemm_bf16_wmma(const bf16* __restrict__ A, const bf16* __restrict__ Bm,
                    float* __restrict__ C,
                    int Md, int Nd, int Kd,
                    int lda, int ldb, int ldc,
                    int64_t sAb, int64_t sAh, int64_t sBb, int64_t sBh,
                    int64_t sCb, int64_t sCh,
                    int Hcnt, float alpha)
{
    __shared__ alignas(16) bf16 As[2][GBM * GBK];
    __shared__ alignas(16) bf16 Bs[2][GBN * GBK];

    const int z  = blockIdx.z;
    const int zb = z / Hcnt;
    const int zh = z - zb * Hcnt;
    A  += (int64_t)zb * sAb + (int64_t)zh * sAh;
    Bm += (int64_t)zb * sBb + (int64_t)zh * sBh;
    C  += (int64_t)zb * sCb + (int64_t)zh * sCh;

    const int m0  = blockIdx.y * GBM;
    const int n0  = blockIdx.x * GBN;
    const int tid = threadIdx.x;
    const int wave = tid >> 5;
    const int lane = tid & 31;
    const int wm = wave & 3;        // 4 wave rows (64 rows each)
    const int wn = wave >> 2;       // 2 wave cols (32 cols each)
    const int lane15   = lane & 15;
    const int laneHalf = lane >> 4;

    // issue async stage of one K-slab into buffer `buf`
    auto stage = [&](int k0, int buf) {
        // A tile: GBM*GBK bf16 = 1024 x 16B chunks -> 4 per thread
#pragma unroll
        for (int it = 0; it < (GBM * GBK / 8) / 256; ++it) {
            int u = tid + it * 256;
            int r = u >> 2, kp = u & 3;
            async_ld_b128(lds_off32(&As[buf][r * GBK + kp * 8]),
                          &A[(size_t)(m0 + r) * lda + k0 + kp * 8]);
        }
        // B tile: GBN*GBK bf16 = 256 x 16B chunks -> 1 per thread
        {
            int u = tid;
            int n = u >> 2, kp = u & 3;
            async_ld_b128(lds_off32(&Bs[buf][n * GBK + kp * 8]),
                          &Bm[(size_t)(n0 + n) * ldb + k0 + kp * 8]);
        }
    };

    v8f acc[4][2] = {};
    stage(0, 0);
    int cur = 0;

    for (int k0 = 0; k0 < Kd; k0 += GBK) {
        wait_asynccnt0();      // my async writes into As/Bs[cur] done
        __syncthreads();       // everyone's done; prev readers of [cur^1] done too
        if (k0 + GBK < Kd) stage(k0 + GBK, cur ^ 1);

        // ---- fragments (CDNA5 16-bit A/B lane layouts, ISA 7.12.2) ----
        v16bf afrag[4], bfrag[2];
#pragma unroll
        for (int i = 0; i < 4; ++i) {
            int m = wm * 64 + i * 16 + lane15;    // tile-local row
            int ka = laneHalf * 8;                // K 0-7 / 8-15 (+16 hi half)
            union { v16bf v; uint4 q[2]; } fa;
            fa.q[0] = *(const uint4*)&As[cur][m * GBK + ka];
            fa.q[1] = *(const uint4*)&As[cur][m * GBK + ka + 16];
            afrag[i] = fa.v;
        }
#pragma unroll
        for (int j = 0; j < 2; ++j) {
            int n = wn * 32 + j * 16 + lane15;    // tile-local col
            int kb = laneHalf * 8;                // K 0-7 / 8-15 (+16 hi half)
            union { v16bf v; uint4 q[2]; } fb;
            fb.q[0] = *(const uint4*)&Bs[cur][n * GBK + kb];
            fb.q[1] = *(const uint4*)&Bs[cur][n * GBK + kb + 16];
            bfrag[j] = fb.v;
        }
#pragma unroll
        for (int i = 0; i < 4; ++i)
#pragma unroll
            for (int j = 0; j < 2; ++j)
                acc[i][j] = __builtin_amdgcn_wmma_f32_16x16x32_bf16(
                    false, afrag[i], false, bfrag[j],
                    (short)0, acc[i][j], false, false);

        __syncthreads();       // frag reads of [cur] done before it is refilled
        cur ^= 1;
    }

    // ---- store: f32 C/D layout: VGPR r -> M = r (+8 for lanes 16-31), N = lane&15 ----
#pragma unroll
    for (int i = 0; i < 4; ++i) {
        int mrow = m0 + wm * 64 + i * 16 + laneHalf * 8;
#pragma unroll
        for (int j = 0; j < 2; ++j) {
            int col = n0 + wn * 32 + j * 16 + lane15;
#pragma unroll
            for (int r = 0; r < 8; ++r)
                C[(size_t)(mrow + r) * ldc + col] = alpha * acc[i][j][r];
        }
    }
}

// ---------------- elementwise kernels ----------------
__global__ void cvt_bf_k(const float* __restrict__ in, bf16* __restrict__ out, size_t n) {
    size_t i = (size_t)blockIdx.x * 256 + threadIdx.x;
    if (i < n) out[i] = (bf16)in[i];
}

// f32 [K,N] row-major -> bf16 [N,K] row-major (coalesced via LDS 32x32 tile)
__global__ __launch_bounds__(256)
void cvt_tr_k(const float* __restrict__ in, bf16* __restrict__ out, int K, int N) {
    __shared__ float tile[32][33];
    int n0 = blockIdx.x * 32;
    int k0 = blockIdx.y * 32;
    int tx = threadIdx.x & 31;
    int ty = threadIdx.x >> 5;          // 0..7
#pragma unroll
    for (int j = 0; j < 32; j += 8)
        tile[ty + j][tx] = in[(size_t)(k0 + ty + j) * N + n0 + tx];
    __syncthreads();
#pragma unroll
    for (int j = 0; j < 32; j += 8)
        out[(size_t)(n0 + ty + j) * K + k0 + tx] = (bf16)tile[tx][ty + j];
}

__global__ void embed_k(const int* __restrict__ inp, const float* __restrict__ emb,
                        float* __restrict__ x) {
    size_t i = (size_t)blockIdx.x * 256 + threadIdx.x;       // S*B*D
    if (i >= (size_t)cS * cB * cD) return;
    int d  = (int)(i % cD);
    size_t sb = i / cD;
    x[i] = emb[(size_t)inp[sb] * cD + d];
}

__global__ void posemb_k(bf16* __restrict__ pe) {
    size_t i = (size_t)blockIdx.x * 256 + threadIdx.x;       // T*D
    if (i >= (size_t)cT * cD) return;
    int d = (int)(i % cD);
    int t = (int)(i / cD);
    float pos = (float)(cT - 1 - t);
    int j = (d < cD / 2) ? d : d - cD / 2;
    float invf = __powf(10000.f, -2.f * (float)j / (float)cD);
    float a = pos * invf;
    pe[i] = (bf16)((d < cD / 2) ? __sinf(a) : __cosf(a));
}

__global__ void build_c_k(const float* __restrict__ mem_l, const float* __restrict__ x,
                          bf16* __restrict__ c) {
    size_t i = (size_t)blockIdx.x * 256 + threadIdx.x;       // T*B*D
    if (i >= (size_t)cT * cB * cD) return;
    size_t half = (size_t)cM * cB * cD;
    c[i] = (bf16)(i < half ? mem_l[i] : x[i - half]);
}

// split fused qkv; q gets pos_bias_u/v added; V stored transposed per (b,h): [B,H,Dh,T]
__global__ void split_qkv_k(const float* __restrict__ qkv,
                            const float* __restrict__ pbu, const float* __restrict__ pbv,
                            bf16* __restrict__ qu, bf16* __restrict__ qv,
                            bf16* __restrict__ kb, bf16* __restrict__ vT) {
    size_t i = (size_t)blockIdx.x * 256 + threadIdx.x;       // T*B*HD
    if (i >= (size_t)cT * cB * cHD) return;
    size_t row = i / cHD;                  // t*B + b
    int c = (int)(i % cHD);                // h*Dh + d
    int t = (int)(row / cB);
    int b = (int)(row % cB);
    int h = c / cDh;
    int d = c % cDh;
    const float* qr = qkv + row * cQ;
    kb[i] = (bf16)qr[cHD + c];
    vT[(((size_t)b * cH + h) * cDh + d) * cT + t] = (bf16)qr[2 * cHD + c];
    if (row >= (size_t)cM * cB) {
        size_t qi = (row - (size_t)cM * cB) * cHD + c;
        float q = qr[c];
        qu[qi] = (bf16)(q + pbu[c]);
        qv[qi] = (bf16)(q + pbv[c]);
    }
}

// TXL rel-shift of BD + add AC + scale + causal mask, in-place into ac.
__global__ void shift_mask_k(float* __restrict__ ac, const float* __restrict__ bd,
                             float scale) {
    size_t i = (size_t)blockIdx.x * 256 + threadIdx.x;       // B*H*S*T
    if (i >= (size_t)cB * cH * cS * cT) return;
    int t = (int)(i % cT);
    size_t r = i / cT;
    int s = (int)(r % cS);
    size_t bh = r / cS;
    const float* bdb = bd + bh * (size_t)cS * cT;
    size_t p = (size_t)cS + (size_t)s * cT + t;              // flat index into padded array
    int s2 = (int)(p / (cT + 1));
    int j  = (int)(p % (cT + 1));
    float bdv = (j > 0) ? bdb[(size_t)s2 * cT + (j - 1)] : 0.f;
    float v = (ac[i] + bdv) * scale;
    ac[i] = (t > s + cM) ? -INFINITY : v;
}

__global__ __launch_bounds__(256)
void softmax_rows_k(const float* __restrict__ s, bf16* __restrict__ p) {
    __shared__ float red[256];
    size_t row = blockIdx.x;                                  // B*H*S rows
    const float* sr = s + row * cT;
    bf16* pr = p + row * cT;
    int tid = threadIdx.x;
    float mx = -INFINITY;
    for (int t = tid; t < cT; t += 256) mx = fmaxf(mx, sr[t]);
    red[tid] = mx; __syncthreads();
    for (int o = 128; o > 0; o >>= 1) {
        if (tid < o) red[tid] = fmaxf(red[tid], red[tid + o]);
        __syncthreads();
    }
    mx = red[0]; __syncthreads();
    float sum = 0.f;
    for (int t = tid; t < cT; t += 256) sum += __expf(sr[t] - mx);
    red[tid] = sum; __syncthreads();
    for (int o = 128; o > 0; o >>= 1) {
        if (tid < o) red[tid] += red[tid + o];
        __syncthreads();
    }
    float inv = 1.f / red[0];
    for (int t = tid; t < cT; t += 256) pr[t] = (bf16)(__expf(sr[t] - mx) * inv);
}

__global__ void bias_relu_k(const float* __restrict__ h, const float* __restrict__ b1,
                            bf16* __restrict__ hb) {
    size_t i = (size_t)blockIdx.x * 256 + threadIdx.x;       // (S*B)*F
    if (i >= (size_t)cS * cB * cF) return;
    int c = (int)(i % cF);
    hb[i] = (bf16)fmaxf(h[i] + b1[c], 0.f);
}

// x = LN(x + add + optional bias), write f32 x and bf16 copy. One block per row.
__global__ __launch_bounds__(256)
void resid_ln_k(float* __restrict__ x, const float* __restrict__ add,
                const float* __restrict__ bias,
                const float* __restrict__ g, const float* __restrict__ bb,
                bf16* __restrict__ xb) {
    __shared__ float red[256];
    size_t row = blockIdx.x;                                  // S*B rows
    int tid = threadIdx.x;
    float loc[cD / 256];
    float s = 0.f;
#pragma unroll
    for (int i = 0; i < cD / 256; ++i) {
        int d = tid + i * 256;
        float v = x[row * cD + d] + add[row * cD + d] + (bias ? bias[d] : 0.f);
        loc[i] = v; s += v;
    }
    red[tid] = s; __syncthreads();
    for (int o = 128; o > 0; o >>= 1) { if (tid < o) red[tid] += red[tid + o]; __syncthreads(); }
    float mean = red[0] / (float)cD; __syncthreads();
    float vs = 0.f;
#pragma unroll
    for (int i = 0; i < cD / 256; ++i) { float d0 = loc[i] - mean; vs += d0 * d0; }
    red[tid] = vs; __syncthreads();
    for (int o = 128; o > 0; o >>= 1) { if (tid < o) red[tid] += red[tid + o]; __syncthreads(); }
    float rn = rsqrtf(red[0] / (float)cD + cEPS);
#pragma unroll
    for (int i = 0; i < cD / 256; ++i) {
        int d = tid + i * 256;
        float y = (loc[i] - mean) * rn * g[d] + bb[d];
        x[row * cD + d] = y;
        xb[row * cD + d] = (bf16)y;
    }
}

// ---------------- host ----------------
static inline unsigned nblk(size_t n) { return (unsigned)((n + 255) / 256); }

extern "C" void kernel_launch(void* const* d_in, const int* in_sizes, int n_in,
                              void* d_out, int out_size, void* d_ws, size_t ws_size,
                              hipStream_t stream) {
    (void)in_sizes; (void)n_in; (void)out_size; (void)ws_size;

    const int*   inputs = (const int*)d_in[0];
    const float* memory = (const float*)d_in[1];
    const float* emb    = (const float*)d_in[2];
    const float* W_qkv  = (const float*)d_in[3];
    const float* W_o    = (const float*)d_in[4];
    const float* W_rel  = (const float*)d_in[5];
    const float* ln1_g  = (const float*)d_in[6];
    const float* ln1_b  = (const float*)d_in[7];
    const float* W_ff1  = (const float*)d_in[8];
    const float* b_ff1  = (const float*)d_in[9];
    const float* W_ff2  = (const float*)d_in[10];
    const float* b_ff2  = (const float*)d_in[11];
    const float* ln2_g  = (const float*)d_in[12];
    const float* ln2_b  = (const float*)d_in[13];
    const float* pbu    = (const float*)d_in[14];
    const float* pbv    = (const float*)d_in[15];
    const float* dec_W  = (const float*)d_in[16];
    float* out = (float*)d_out;

    // deterministic bump allocator inside d_ws
    char* base = (char*)d_ws;
    size_t off = 0;
    auto alloc = [&](size_t bytes) -> char* {
        off = (off + 255) & ~(size_t)255;
        char* p = base + off;
        off += bytes;
        return p;
    };
    const size_t SB  = (size_t)cS * cB;     // 2048
    const size_t TBr = (size_t)cT * cB;     // 4096

    float* xf    = (float*)alloc(SB * cD * 4);
    float* qkvf  = (float*)alloc(TBr * cQ * 4);
    float* rf    = (float*)alloc((size_t)cT * cHD * 4);
    float* acbuf = (float*)alloc((size_t)cB * cH * cS * cT * 4);
    float* bdbuf = (float*)alloc((size_t)cB * cH * cS * cT * 4);
    float* vecf  = (float*)alloc(SB * cHD * 4);
    float* attnf = (float*)alloc(SB * cD * 4);
    float* hf    = (float*)alloc(SB * cF * 4);
    bf16* xbf    = (bf16*)alloc(SB * cD * 2);
    bf16* cbf    = (bf16*)alloc(TBr * cD * 2);
    bf16* qu     = (bf16*)alloc(SB * cHD * 2);
    bf16* qv     = (bf16*)alloc(SB * cHD * 2);
    bf16* kbf    = (bf16*)alloc(TBr * cHD * 2);
    bf16* vT     = (bf16*)alloc((size_t)cB * cH * cDh * cT * 2);
    bf16* rbf    = (bf16*)alloc((size_t)cT * cHD * 2);
    bf16* posbf  = (bf16*)alloc((size_t)cT * cD * 2);
    bf16* prob   = (bf16*)alloc((size_t)cB * cH * cS * cT * 2);
    bf16* vecbf  = (bf16*)alloc(SB * cHD * 2);
    bf16* hbf    = (bf16*)alloc(SB * cF * 2);
    bf16* wqkvT  = (bf16*)alloc((size_t)cQ * cD * 2);   // [3072,1024]
    bf16* woT    = (bf16*)alloc((size_t)cD * cHD * 2);  // [1024,1024]
    bf16* wrelT  = (bf16*)alloc((size_t)cHD * cD * 2);  // [1024,1024]
    bf16* wff1T  = (bf16*)alloc((size_t)cF * cD * 2);   // [4096,1024]
    bf16* wff2T  = (bf16*)alloc((size_t)cD * cF * 2);   // [1024,4096]
    bf16* wdec   = (bf16*)acbuf;   // alias: scores dead by decode time; dec_W already [V,D]=[N,K]

    auto gemm = [&](const bf16* A, const bf16* Bm, float* C,
                    int Md, int Nd, int Kd, int lda, int ldb, int ldc,
                    int64_t sAb, int64_t sAh, int64_t sBb, int64_t sBh,
                    int64_t sCb, int64_t sCh,
                    int Hcnt, int nbatch, float alpha) {
        dim3 g((unsigned)(Nd / GBN), (unsigned)(Md / GBM), (unsigned)nbatch);
        gemm_bf16_wmma<<<g, dim3(256), 0, stream>>>(A, Bm, C, Md, Nd, Kd,
                                                    lda, ldb, ldc,
                                                    sAb, sAh, sBb, sBh, sCb, sCh,
                                                    Hcnt, alpha);
    };
    auto cvt = [&](const float* in, bf16* o, size_t n) {
        cvt_bf_k<<<nblk(n), 256, 0, stream>>>(in, o, n);
    };
    auto cvtT = [&](const float* in, bf16* o, int K, int N) {
        dim3 g((unsigned)(N / 32), (unsigned)(K / 32));
        cvt_tr_k<<<g, dim3(256), 0, stream>>>(in, o, K, N);
    };

    // ---- prologue ----
    embed_k<<<nblk(SB * cD), 256, 0, stream>>>(inputs, emb, xf);
    posemb_k<<<nblk((size_t)cT * cD), 256, 0, stream>>>(posbf);

    const float scale = 0.125f;  // 1/sqrt(64)
    const int64_t sST = (int64_t)cS * cT;

    for (int l = 0; l < cL; ++l) {
        // per-layer weight conversion (transpose to [N,K] bf16)
        cvtT(W_qkv + (size_t)l * cD * cQ,  wqkvT, cD, cQ);
        cvtT(W_o   + (size_t)l * cHD * cD, woT,   cHD, cD);
        cvtT(W_rel + (size_t)l * cD * cHD, wrelT, cD, cHD);
        cvtT(W_ff1 + (size_t)l * cD * cF,  wff1T, cD, cF);
        cvtT(W_ff2 + (size_t)l * cF * cD,  wff2T, cF, cD);

        // c = concat(memory[l], x)  -> bf16
        build_c_k<<<nblk(TBr * cD), 256, 0, stream>>>(memory + (size_t)l * cM * cB * cD, xf, cbf);

        // qkv = c @ W_qkv  [4096 x 3072]
        gemm(cbf, wqkvT, qkvf, (int)TBr, cQ, cD, cD, cD, cQ, 0,0,0,0,0,0, 1, 1, 1.f);
        split_qkv_k<<<nblk(TBr * cHD), 256, 0, stream>>>(qkvf, pbu, pbv, qu, qv, kbf, vT);

        // r = pos_emb @ W_rel  [1024 x 1024]
        gemm(posbf, wrelT, rf, cT, cHD, cD, cD, cD, cHD, 0,0,0,0,0,0, 1, 1, 1.f);
        cvt(rf, rbf, (size_t)cT * cHD);

        // AC[b,h] = (q+u) @ K^T   M=512 N=1024 K=64, batched over 64 (b,h)
        gemm(qu, kbf, acbuf, cS, cT, cDh,
             cB * cHD, cB * cHD, cT,
             cHD, cDh, cHD, cDh, (int64_t)cH * sST, sST,
             cH, cB * cH, 1.f);
        // BD[b,h] = (q+v) @ r_h^T   (rbf rows are [T, HD], per-head base h*Dh)
        gemm(qv, rbf, bdbuf, cS, cT, cDh,
             cB * cHD, cHD, cT,
             cHD, cDh, 0, cDh, (int64_t)cH * sST, sST,
             cH, cB * cH, 1.f);

        shift_mask_k<<<nblk((size_t)cB * cH * cS * cT), 256, 0, stream>>>(acbuf, bdbuf, scale);
        softmax_rows_k<<<(unsigned)((size_t)cB * cH * cS), 256, 0, stream>>>(acbuf, prob);

        // vec[b,h] = prob @ V   M=512 N=64 K=1024 ; V pre-transposed [B,H,Dh,T]
        gemm(prob, vT, vecf, cS, cDh, cT,
             cT, cT, cB * cHD,
             (int64_t)cH * sST, sST,
             (int64_t)cH * cDh * cT, (int64_t)cDh * cT,
             cHD, cDh,
             cH, cB * cH, 1.f);
        cvt(vecf, vecbf, SB * cHD);

        // attn_out = vec @ W_o
        gemm(vecbf, woT, attnf, (int)SB, cD, cHD, cHD, cHD, cD, 0,0,0,0,0,0, 1, 1, 1.f);
        resid_ln_k<<<(unsigned)SB, 256, 0, stream>>>(xf, attnf, nullptr,
                                                     ln1_g + (size_t)l * cD, ln1_b + (size_t)l * cD, xbf);

        // FF
        gemm(xbf, wff1T, hf, (int)SB, cF, cD, cD, cD, cF, 0,0,0,0,0,0, 1, 1, 1.f);
        bias_relu_k<<<nblk(SB * cF), 256, 0, stream>>>(hf, b_ff1 + (size_t)l * cF, hbf);
        gemm(hbf, wff2T, attnf, (int)SB, cD, cF, cF, cF, cD, 0,0,0,0,0,0, 1, 1, 1.f);
        resid_ln_k<<<(unsigned)SB, 256, 0, stream>>>(xf, attnf, b_ff2 + (size_t)l * cD,
                                                     ln2_g + (size_t)l * cD, ln2_b + (size_t)l * cD, xbf);
    }

    // decode: logits = x @ dec_W^T   M=2048 N=32000 K=1024 (dec_W already [N,K])
    cvt(dec_W, wdec, (size_t)cV * cD);
    gemm(xbf, wdec, out, (int)SB, cV, cD, cD, cD, cV, 0,0,0,0,0,0, 1, 1, 1.f);
}
